// _SAModuleSSG_68436008895057
// MI455X (gfx1250) — matrix operand order repaired and checked by reading, compiled, not run on hardware
//
#include <hip/hip_runtime.h>

typedef __attribute__((ext_vector_type(16))) _Float16 v16h;
typedef __attribute__((ext_vector_type(8)))  _Float16 v8h;
typedef __attribute__((ext_vector_type(8)))  float    v8f;

#define B_   8
#define N_   8192
#define C_   64
#define S_   2048
#define NS_  32
#define P_   (B_ * S_ * NS_)      // 524288 positions
#define RAD2 0.04f
#define EPS_ 1e-5f

// ---------------------------------------------------------------------------
// 1) Furthest point sampling: one block per batch, dists live in LDS.
// ---------------------------------------------------------------------------
__global__ void __launch_bounds__(1024) fps_kernel(const float* __restrict__ xyz,
                                                   int* __restrict__ fps_idx,
                                                   float* __restrict__ new_xyz) {
  __shared__ float dists[N_];       // 32 KB
  __shared__ float rv[1024];
  __shared__ int   ri[1024];
  __shared__ int   s_last;
  __shared__ float lp[3];
  const int b = blockIdx.x;
  const int t = threadIdx.x;
  const float* X = xyz + (size_t)b * N_ * 3;

  for (int j = t; j < N_; j += 1024) dists[j] = 1e10f;
  if (t == 0) {
    s_last = 0;
    fps_idx[b * S_ + 0] = 0;
    new_xyz[((size_t)b * S_ + 0) * 3 + 0] = X[0];
    new_xyz[((size_t)b * S_ + 0) * 3 + 1] = X[1];
    new_xyz[((size_t)b * S_ + 0) * 3 + 2] = X[2];
  }
  __syncthreads();

  for (int i = 1; i < S_; ++i) {
    if (t < 3) lp[t] = X[s_last * 3 + t];
    __syncthreads();
    const float lx = lp[0], ly = lp[1], lz = lp[2];
    float bv = -1.0f; int bi = 0x7fffffff;
    for (int j = t; j < N_; j += 1024) {
      float dx = X[j * 3 + 0] - lx;
      float dy = X[j * 3 + 1] - ly;
      float dz = X[j * 3 + 2] - lz;
      float d  = dx * dx + dy * dy + dz * dz;
      float nd = fminf(dists[j], d);
      dists[j] = nd;
      if (nd > bv) { bv = nd; bi = j; }      // j ascending -> first-max kept
    }
    rv[t] = bv; ri[t] = bi;
    __syncthreads();
    for (int s = 512; s > 0; s >>= 1) {
      if (t < s) {
        float ov = rv[t + s]; int oi = ri[t + s];
        if (ov > rv[t] || (ov == rv[t] && oi < ri[t])) { rv[t] = ov; ri[t] = oi; }
      }
      __syncthreads();
    }
    if (t == 0) {
      int nxt = ri[0];
      s_last = nxt;
      fps_idx[b * S_ + i] = nxt;
      new_xyz[((size_t)b * S_ + i) * 3 + 0] = X[nxt * 3 + 0];
      new_xyz[((size_t)b * S_ + i) * 3 + 1] = X[nxt * 3 + 1];
      new_xyz[((size_t)b * S_ + i) * 3 + 2] = X[nxt * 3 + 2];
    }
    __syncthreads();
  }
}

// ---------------------------------------------------------------------------
// 2) Ball query: ordered scan == reference top_k(-key) semantics exactly.
// ---------------------------------------------------------------------------
__global__ void ballquery_kernel(const float* __restrict__ xyz,
                                 const float* __restrict__ new_xyz,
                                 int* __restrict__ idx) {
  int q = blockIdx.x * blockDim.x + threadIdx.x;
  if (q >= B_ * S_) return;
  int b = q / S_;
  const float* X = xyz + (size_t)b * N_ * 3;
  float qx = new_xyz[(size_t)q * 3 + 0];
  float qy = new_xyz[(size_t)q * 3 + 1];
  float qz = new_xyz[(size_t)q * 3 + 2];
  int* o = idx + (size_t)q * NS_;
  int cnt = 0, first = 0;
  for (int j = 0; j < N_ && cnt < NS_; ++j) {
    float dx = X[j * 3 + 0] - qx;
    float dy = X[j * 3 + 1] - qy;
    float dz = X[j * 3 + 2] - qz;
    if (dx * dx + dy * dy + dz * dz < RAD2) {
      if (cnt == 0) first = j;
      o[cnt++] = j;
    }
  }
  for (; cnt < NS_; ++cnt) o[cnt] = first;
}

// ---------------------------------------------------------------------------
// 3) Transpose features (B,C,N)->(B,N,C) as f16, LDS-tiled.
// ---------------------------------------------------------------------------
__global__ void __launch_bounds__(256) tfeat_kernel(const float* __restrict__ F,
                                                    _Float16* __restrict__ FT) {
  __shared__ float tile[32][33];
  int blk   = blockIdx.x;
  int ctile = blk % (C_ / 32);
  int jtile = (blk / (C_ / 32)) % (N_ / 32);
  int b     = blk / ((C_ / 32) * (N_ / 32));
  int c0 = ctile * 32, j0 = jtile * 32;
  int tx = threadIdx.x & 31, ty = threadIdx.x >> 5;
  for (int pass = 0; pass < 4; ++pass) {
    int cl = ty + pass * 8;
    tile[cl][tx] = F[((size_t)b * C_ + c0 + cl) * N_ + j0 + tx];
  }
  __syncthreads();
  for (int pass = 0; pass < 4; ++pass) {
    int row = ty + pass * 8;
    FT[((size_t)b * N_ + j0 + row) * C_ + c0 + tx] = (_Float16)tile[tx][row];
  }
}

// ---------------------------------------------------------------------------
// 4) Weights -> f16 (W1 padded 67 -> 96 cols with zeros).
// ---------------------------------------------------------------------------
__global__ void prep_weights_kernel(const float* __restrict__ W1,
                                    const float* __restrict__ W2,
                                    const float* __restrict__ W3,
                                    _Float16* __restrict__ W1f,
                                    _Float16* __restrict__ W2f,
                                    _Float16* __restrict__ W3f) {
  int i = blockIdx.x * blockDim.x + threadIdx.x;
  if (i < 64 * 96) {
    int o = i / 96, c = i % 96;
    W1f[i] = (c < 67) ? (_Float16)W1[o * 67 + c] : (_Float16)0.0f;
  } else if (i < 64 * 96 + 64 * 64) {
    int k = i - 64 * 96;
    W2f[k] = (_Float16)W2[k];
  } else if (i < 64 * 96 + 64 * 64 + 128 * 64) {
    int k = i - (64 * 96 + 64 * 64);
    W3f[k] = (_Float16)W3[k];
  }
}

// ---------------------------------------------------------------------------
// 5) Group: build g1 as (P, 96) f16 rows: [xyz-diff(3) | feat(64) | pad(29)].
//    One wave per position; coalesced f16 gathers and writes.
// ---------------------------------------------------------------------------
__global__ void __launch_bounds__(256) group_kernel(const float* __restrict__ xyz,
                                                    const _Float16* __restrict__ FT,
                                                    const float* __restrict__ new_xyz,
                                                    const int* __restrict__ idx,
                                                    _Float16* __restrict__ G) {
  int p    = blockIdx.x * 8 + (threadIdx.x >> 5);
  int lane = threadIdx.x & 31;
  int n = p % NS_;
  int s = (p / NS_) % S_;
  int b = p / (NS_ * S_);
  int q = b * S_ + s;
  int j = idx[(size_t)q * NS_ + n];
  j = (j < N_ - 1) ? j : (N_ - 1);
  const _Float16* ft = FT + ((size_t)b * N_ + j) * C_;
  _Float16* g = G + (size_t)p * 96;
  for (int c = lane; c < 96; c += 32) {
    float v;
    if (c < 3)       v = xyz[((size_t)b * N_ + j) * 3 + c] - new_xyz[(size_t)q * 3 + c];
    else if (c < 67) v = (float)ft[c - 3];
    else             v = 0.0f;
    g[c] = (_Float16)v;
  }
}

// ---------------------------------------------------------------------------
// 6) WMMA GEMM  H(o,p) = Wf(o,c) * G(p,c)^T  + per-channel sum / sumsq stats.
//    One wave per 16x16 output tile; v_wmma_f32_16x16x32_f16.
// ---------------------------------------------------------------------------
__global__ void __launch_bounds__(256) gemm_kernel(const _Float16* __restrict__ A,
                                                   const _Float16* __restrict__ Bm,
                                                   float* __restrict__ H,
                                                   float* __restrict__ sum,
                                                   float* __restrict__ sumsq,
                                                   int Cpad) {
  __shared__ float ls_sum[16];
  __shared__ float ls_sq[16];
  const int t = threadIdx.x;
  if (t < 16) { ls_sum[t] = 0.0f; ls_sq[t] = 0.0f; }
  __syncthreads();

  const int wave = t >> 5;
  const int lane = t & 31;
  const int half = lane >> 4;     // K-half select
  const int l16  = lane & 15;     // M for A, N for B
  const int otile = blockIdx.y;
  const int p0    = (blockIdx.x * 8 + wave) * 16;

  const size_t rowA = (size_t)(otile * 16 + l16) * Cpad;
  const size_t rowB = (size_t)(p0 + l16) * Cpad;

  v8f acc = {};
  for (int k0 = 0; k0 < Cpad; k0 += 32) {
    // A 16x32: lanes 0-15 K=0..15 ; lanes 16-31 K=16..31 (contiguous 32B)
    v16h a  = *(const v16h*)(A + rowA + k0 + half * 16);
    // B 32x16: lanes 0-15 K=0..15 ; lanes 16-31 K=16..31 (contiguous 32B)
    v16h bm = *(const v16h*)(Bm + rowB + k0 + half * 16);
    acc = __builtin_amdgcn_wmma_f32_16x16x32_f16(false, a, false, bm,
                                                 (short)0, acc, false, false);
  }

  // D layout: VGPR r -> M = r + 8*half ; N = l16
  for (int r = 0; r < 8; ++r) {
    int chl = r + half * 8;
    int ch  = otile * 16 + chl;
    float v = acc[r];
    H[(size_t)ch * P_ + p0 + l16] = v;
    float sv = v, sq = v * v;
    for (int m = 1; m < 16; m <<= 1) {
      sv += __shfl_xor(sv, m, 32);
      sq += __shfl_xor(sq, m, 32);
    }
    if (l16 == 0) {
      atomicAdd(&ls_sum[chl], sv);   // ds_add_f32
      atomicAdd(&ls_sq[chl], sq);
    }
  }
  __syncthreads();
  if (t < 16) {
    atomicAdd(&sum[otile * 16 + t], ls_sum[t]);
    atomicAdd(&sumsq[otile * 16 + t], ls_sq[t]);
  }
}

// ---------------------------------------------------------------------------
// 7) Per-channel BN coefficients from the global stats.
// ---------------------------------------------------------------------------
__global__ void finalize_kernel(const float* __restrict__ sum,
                                const float* __restrict__ sumsq,
                                const float* __restrict__ gamma,
                                const float* __restrict__ beta,
                                float* __restrict__ scale,
                                float* __restrict__ shift, int O) {
  int c = blockIdx.x * blockDim.x + threadIdx.x;
  if (c >= O) return;
  float mu  = sum[c] * (1.0f / (float)P_);
  float var = sumsq[c] * (1.0f / (float)P_) - mu * mu;
  float inv = rsqrtf(var + EPS_);
  float sc  = gamma[c] * inv;
  scale[c] = sc;
  shift[c] = beta[c] - mu * sc;
}

// ---------------------------------------------------------------------------
// 8) BN + ReLU + transpose-repack H(o,p) f32 -> G(p,64) f16 (LDS-tiled).
// ---------------------------------------------------------------------------
__global__ void __launch_bounds__(256) bnpack_kernel(const float* __restrict__ H,
                                                     const float* __restrict__ scale,
                                                     const float* __restrict__ shift,
                                                     _Float16* __restrict__ G) {
  __shared__ _Float16 tile[64][33];
  const int t  = threadIdx.x;
  const int p0 = blockIdx.x * 32;
  {
    int pl = t & 31, chb = t >> 5;
    for (int pass = 0; pass < 8; ++pass) {
      int ch = chb + pass * 8;
      float v = H[(size_t)ch * P_ + p0 + pl];
      float y = fmaxf(v * scale[ch] + shift[ch], 0.0f);
      tile[ch][pl] = (_Float16)y;
    }
  }
  __syncthreads();
  {
    int ch = t & 63, plb = t >> 6;
    for (int pass = 0; pass < 8; ++pass) {
      int pl = plb + pass * 4;
      G[((size_t)(p0 + pl)) * 64 + ch] = tile[ch][pl];
    }
  }
}

// ---------------------------------------------------------------------------
// 9) Layer-3 BN + ReLU + max over the 32 samples -> feats (B,128,S).
// ---------------------------------------------------------------------------
__global__ void finalmax_kernel(const float* __restrict__ H,
                                const float* __restrict__ scale,
                                const float* __restrict__ shift,
                                float* __restrict__ out) {
  int t = blockIdx.x * blockDim.x + threadIdx.x;
  if (t >= B_ * 128 * S_) return;
  int s  = t % S_;
  int ch = (t / S_) % 128;
  int b  = t / (S_ * 128);
  size_t p0 = ((size_t)b * S_ + s) * NS_;
  float sc = scale[ch], sh = shift[ch];
  const float* h = H + (size_t)ch * P_ + p0;
  float m = 0.0f;                       // ReLU outputs are >= 0
  for (int n = 0; n < NS_; ++n) m = fmaxf(m, fmaxf(h[n] * sc + sh, 0.0f));
  out[(size_t)(b * 128 + ch) * S_ + s] = m;
}

// ---------------------------------------------------------------------------
extern "C" void kernel_launch(void* const* d_in, const int* in_sizes, int n_in,
                              void* d_out, int out_size, void* d_ws, size_t ws_size,
                              hipStream_t stream) {
  (void)in_sizes; (void)n_in; (void)out_size; (void)ws_size;
  const float* xyz      = (const float*)d_in[0];
  const float* features = (const float*)d_in[1];
  const float* W1  = (const float*)d_in[2];
  const float* ga1 = (const float*)d_in[3];
  const float* be1 = (const float*)d_in[4];
  const float* W2  = (const float*)d_in[5];
  const float* ga2 = (const float*)d_in[6];
  const float* be2 = (const float*)d_in[7];
  const float* W3  = (const float*)d_in[8];
  const float* ga3 = (const float*)d_in[9];
  const float* be3 = (const float*)d_in[10];
  float* out = (float*)d_out;
  char*  ws  = (char*)d_ws;

  int*      fps_idx = (int*)(ws + 0);                    //   64 KB
  int*      idx     = (int*)(ws + 65536);                //    2 MB
  _Float16* FT      = (_Float16*)(ws + 2162688);         //    8 MB
  _Float16* W1f     = (_Float16*)(ws + 10551296);
  _Float16* W2f     = (_Float16*)(ws + 10563584);
  _Float16* W3f     = (_Float16*)(ws + 10571776);
  float*    sum     = (float*)(ws + 10588160);
  float*    sumsq   = (float*)(ws + 10588160 + 512);
  float*    scale   = (float*)(ws + 10588160 + 1024);
  float*    shift   = (float*)(ws + 10588160 + 1536);
  _Float16* G       = (_Float16*)(ws + 10590208);        // ~100 MB (P x 96 f16)
  float*    H       = (float*)(ws + 111253504);          // ~268 MB (128 x P f32)

  float* new_xyz = out;                 // (B,S,3) = first 49152 floats
  float* feats   = out + B_ * S_ * 3;   // (B,128,S)

  fps_kernel<<<B_, 1024, 0, stream>>>(xyz, fps_idx, new_xyz);
  tfeat_kernel<<<B_ * (N_ / 32) * (C_ / 32), 256, 0, stream>>>(features, FT);
  prep_weights_kernel<<<(64 * 96 + 64 * 64 + 128 * 64 + 255) / 256, 256, 0, stream>>>(
      W1, W2, W3, W1f, W2f, W3f);
  ballquery_kernel<<<(B_ * S_ + 255) / 256, 256, 0, stream>>>(xyz, new_xyz, idx);
  group_kernel<<<P_ / 8, 256, 0, stream>>>(xyz, FT, new_xyz, idx, G);

  // layer 1: 64 x 96(pad) GEMM
  hipMemsetAsync(sum, 0, 1024, stream);
  gemm_kernel<<<dim3(P_ / 128, 4), 256, 0, stream>>>(W1f, G, H, sum, sumsq, 96);
  finalize_kernel<<<1, 64, 0, stream>>>(sum, sumsq, ga1, be1, scale, shift, 64);
  bnpack_kernel<<<P_ / 32, 256, 0, stream>>>(H, scale, shift, G);

  // layer 2: 64 x 64 GEMM
  hipMemsetAsync(sum, 0, 1024, stream);
  gemm_kernel<<<dim3(P_ / 128, 4), 256, 0, stream>>>(W2f, G, H, sum, sumsq, 64);
  finalize_kernel<<<1, 64, 0, stream>>>(sum, sumsq, ga2, be2, scale, shift, 64);
  bnpack_kernel<<<P_ / 32, 256, 0, stream>>>(H, scale, shift, G);

  // layer 3: 128 x 64 GEMM + max-pool
  hipMemsetAsync(sum, 0, 1024, stream);
  gemm_kernel<<<dim3(P_ / 128, 8), 256, 0, stream>>>(W3f, G, H, sum, sumsq, 64);
  finalize_kernel<<<1, 128, 0, stream>>>(sum, sumsq, ga3, be3, scale, shift, 128);
  finalmax_kernel<<<(B_ * 128 * S_ + 255) / 256, 256, 0, stream>>>(H, scale, shift, feats);
}
